// GWNN2_41970420418156
// MI455X (gfx1250) — matrix operand order repaired and checked by reading, not compiled, not run on hardware
//
#include <hip/hip_runtime.h>
#include <hip/hip_bf16.h>
#include <cstdint>
#include <cstddef>

// ---- CDNA5 WMMA vector types ----
typedef __attribute__((ext_vector_type(16))) __bf16 v16bf;
typedef __attribute__((ext_vector_type(8)))  __bf16 v8bf;
typedef __attribute__((ext_vector_type(8)))  float  v8f;

#define N_NODES   50000
#define G_NUM     4
#define N_EDGES   800000
#define IN_FEATS  256
#define H_FEATS   128
#define NUM_CLASSES 40
#define CAT       512   // H_FEATS * G_NUM

__device__ __forceinline__ __bf16 f2bf(float f) {
  unsigned u = __float_as_uint(f);
  u += 0x7FFFu + ((u >> 16) & 1u);           // round-to-nearest-even
  unsigned short h = (unsigned short)(u >> 16);
  return __builtin_bit_cast(__bf16, h);
}

// ---------------- utility kernels ----------------
__global__ void k_zero_f32(float* __restrict__ p, int n) {
  int i = blockIdx.x * blockDim.x + threadIdx.x;
  if (i < n) p[i] = 0.0f;
}

__global__ void k_count_deg(const int* __restrict__ idx, float* __restrict__ deg,
                            int nE, int nNodes) {
  int i = blockIdx.x * blockDim.x + threadIdx.x;
  if (i >= G_NUM * nE) return;
  int g = i / nE;
  int node = idx[i];
  unsafeAtomicAdd(&deg[g * nNodes + node], 1.0f);
}

__global__ void k_rsqrt_clip(float* __restrict__ p, int n) {
  int i = blockIdx.x * blockDim.x + threadIdx.x;
  if (i < n) p[i] = rsqrtf(fmaxf(p[i], 1.0f));
}

__global__ void k_f32_to_bf16(const float* __restrict__ x, __bf16* __restrict__ y, int n) {
  int i = blockIdx.x * blockDim.x + threadIdx.x;
  if (i < n) y[i] = f2bf(x[i]);
}

// W is [K,N] row-major f32; Bt is [Npad,K] row-major bf16 (zero padded rows).
__global__ void k_transpose_bf16(const float* __restrict__ W, __bf16* __restrict__ Bt,
                                 int K, int N, int Npad) {
  int i = blockIdx.x * blockDim.x + threadIdx.x;
  if (i >= Npad * K) return;
  int nrow = i / K, k = i - nrow * K;
  float v = (nrow < N) ? W[(size_t)k * N + nrow] : 0.0f;
  Bt[i] = f2bf(v);
}

// ---------------- WMMA GEMM ----------------
// C[M,Nout] = A[M,K](bf16) * Bt[Npad,K]^T(bf16)  (+bias) (relu) ; f32 accumulate.
// One wave computes a 16(M) x 64(N) slab: one A fragment reused over 4 WMMAs.
// M must be a multiple of 16 (50000 = 3125*16). K multiple of 32.
template <bool OUT_BF16, bool RELU, bool BIAS>
__global__ __launch_bounds__(256) void k_gemm_wmma(
    const __bf16* __restrict__ A, const __bf16* __restrict__ Bt,
    const float* __restrict__ bias, void* __restrict__ Cout,
    int M, int K, int Nt64, int Nout, int ldc) {
  int wave = (int)((blockIdx.x * blockDim.x + threadIdx.x) >> 5);
  int lane = threadIdx.x & 31;
  int mTiles = M >> 4;
  int mT = wave % mTiles;
  int nT = wave / mTiles;
  if (nT >= Nt64) return;              // wave-uniform exit (EXEC stays full)
  int m0 = mT << 4, n0 = nT << 6;
  int lm = lane & 15, lh = lane >> 4;

  // A lane layout (16x32 bf16): elems 0..15 -> K = 16*lh + 0..15
  const __bf16* arow = A + (size_t)(m0 + lm) * K + lh * 16;
  // B lane layout (32x16 bf16): lane holds column n = lm, K = 16*lh + 0..15 (contiguous in Bt row)
  const __bf16* brow = Bt + (size_t)(n0 + lm) * K + lh * 16;

  v8f acc0 = {}, acc1 = {}, acc2 = {}, acc3 = {};
  for (int k = 0; k < K; k += 32) {
    v8bf alo = *(const v8bf*)(arow + k);
    v8bf ahi = *(const v8bf*)(arow + k + 8);
    v16bf a;
#pragma unroll
    for (int i = 0; i < 8; ++i) { a[i] = alo[i]; a[8 + i] = ahi[i]; }
    v16bf b0 = *(const v16bf*)(brow + k);
    v16bf b1 = *(const v16bf*)(brow + (size_t)16 * K + k);
    v16bf b2 = *(const v16bf*)(brow + (size_t)32 * K + k);
    v16bf b3 = *(const v16bf*)(brow + (size_t)48 * K + k);
    acc0 = __builtin_amdgcn_wmma_f32_16x16x32_bf16(false, a, false, b0, (short)0, acc0, false, false);
    acc1 = __builtin_amdgcn_wmma_f32_16x16x32_bf16(false, a, false, b1, (short)0, acc1, false, false);
    acc2 = __builtin_amdgcn_wmma_f32_16x16x32_bf16(false, a, false, b2, (short)0, acc2, false, false);
    acc3 = __builtin_amdgcn_wmma_f32_16x16x32_bf16(false, a, false, b3, (short)0, acc3, false, false);
  }

  float*  of = (float*)Cout;
  __bf16* ob = (__bf16*)Cout;
  v8f accs[4] = {acc0, acc1, acc2, acc3};
#pragma unroll
  for (int t = 0; t < 4; ++t) {
    int n = n0 + t * 16 + lm;
    if (n >= Nout) continue;
    float bv = BIAS ? bias[n] : 0.0f;
#pragma unroll
    for (int r = 0; r < 8; ++r) {
      // C/D layout: VGPR r -> M = r + 8*lh ; N = lane%16
      int m = m0 + lh * 8 + r;
      float v = accs[t][r] + bv;
      if (RELU) v = fmaxf(v, 0.0f);
      size_t o = (size_t)m * ldc + n;
      if constexpr (OUT_BF16) ob[o] = f2bf(v); else of[o] = v;
    }
  }
}

// ---------------- SpMM: one wave per edge, 128 feats = 32 lanes x float4 ----------------
__global__ __launch_bounds__(256) void k_spmm(
    const int* __restrict__ src, const int* __restrict__ dst,
    const float* __restrict__ w, const float* __restrict__ degout_r,
    const float* __restrict__ Y, float* __restrict__ agg, int nE) {
  int wave = (int)((blockIdx.x * blockDim.x + threadIdx.x) >> 5);
  int lane = threadIdx.x & 31;
  if (wave >= nE) return;
  int s = src[wave], d = dst[wave];
  float coef = w[wave] * degout_r[s];
  const float4* yv = (const float4*)(Y + (size_t)s * H_FEATS);
  float4 v = yv[lane];
  float* a = agg + (size_t)d * H_FEATS + lane * 4;
  unsafeAtomicAdd(a + 0, v.x * coef);
  unsafeAtomicAdd(a + 1, v.y * coef);
  unsafeAtomicAdd(a + 2, v.z * coef);
  unsafeAtomicAdd(a + 3, v.w * coef);
}

// relu(agg * deg_in^-0.5) -> bf16, scattered into the concat activation at column gOff
__global__ void k_finalize(const float* __restrict__ agg, const float* __restrict__ degin_r,
                           __bf16* __restrict__ Cb, int gOff) {
  int i = blockIdx.x * blockDim.x + threadIdx.x;
  if (i >= N_NODES * H_FEATS) return;
  int n = i >> 7, f = i & 127;
  float v = fmaxf(agg[i] * degin_r[n], 0.0f);
  Cb[(size_t)n * CAT + gOff + f] = f2bf(v);
}

// ---------------- host orchestration ----------------
extern "C" void kernel_launch(void* const* d_in, const int* in_sizes, int n_in,
                              void* d_out, int out_size, void* d_ws, size_t ws_size,
                              hipStream_t stream) {
  (void)in_sizes; (void)n_in; (void)out_size; (void)ws_size;
  const float* in_feat = (const float*)d_in[0];
  const int*   src     = (const int*)d_in[1];
  const int*   dst     = (const int*)d_in[2];
  const float* w       = (const float*)d_in[3];
  const float* W1      = (const float*)d_in[4];
  const float* W2      = (const float*)d_in[5];
  const float* l1w     = (const float*)d_in[6];
  const float* l1b     = (const float*)d_in[7];
  const float* l2w     = (const float*)d_in[8];
  const float* l2b     = (const float*)d_in[9];
  const float* l3w     = (const float*)d_in[10];
  const float* l3b     = (const float*)d_in[11];

  char* ws = (char*)d_ws;
  size_t off = 0;
  auto alloc = [&](size_t bytes) -> char* {
    char* p = ws + off;
    off += (bytes + 255) & ~(size_t)255;
    return p;
  };
  float*  degout = (float*)alloc((size_t)G_NUM * N_NODES * 4);
  float*  degin  = (float*)alloc((size_t)G_NUM * N_NODES * 4);
  __bf16* Xb     = (__bf16*)alloc((size_t)N_NODES * IN_FEATS * 2);
  __bf16* W1t    = (__bf16*)alloc((size_t)H_FEATS * IN_FEATS * 2);
  __bf16* l1t    = (__bf16*)alloc((size_t)CAT * CAT * 2);
  __bf16* l2t    = (__bf16*)alloc((size_t)CAT * CAT * 2);
  __bf16* W2t    = (__bf16*)alloc((size_t)H_FEATS * CAT * 2);
  __bf16* l3t    = (__bf16*)alloc((size_t)64 * CAT * 2);       // 40 padded to 64 rows
  float*  Y      = (float*)alloc((size_t)N_NODES * H_FEATS * 4); // Y1, later Z
  float*  agg    = (float*)alloc((size_t)N_NODES * H_FEATS * 4);
  __bf16* bufA   = (__bf16*)alloc((size_t)N_NODES * CAT * 2);  // C1, later H2
  __bf16* bufB   = (__bf16*)alloc((size_t)N_NODES * CAT * 2);  // H1, later C2

  const int T = 256;
  const int ndeg = G_NUM * N_NODES;
  const int tot  = G_NUM * N_EDGES;
  const int nagg = N_NODES * H_FEATS;

  // --- degrees ---
  k_zero_f32<<<(ndeg + T - 1) / T, T, 0, stream>>>(degout, ndeg);
  k_zero_f32<<<(ndeg + T - 1) / T, T, 0, stream>>>(degin, ndeg);
  k_count_deg<<<(tot + T - 1) / T, T, 0, stream>>>(src, degout, N_EDGES, N_NODES);
  k_count_deg<<<(tot + T - 1) / T, T, 0, stream>>>(dst, degin, N_EDGES, N_NODES);
  k_rsqrt_clip<<<(ndeg + T - 1) / T, T, 0, stream>>>(degout, ndeg);
  k_rsqrt_clip<<<(ndeg + T - 1) / T, T, 0, stream>>>(degin, ndeg);

  // --- bf16 converts / weight transposes ---
  const int nx = N_NODES * IN_FEATS;
  k_f32_to_bf16<<<(nx + T - 1) / T, T, 0, stream>>>(in_feat, Xb, nx);
  k_transpose_bf16<<<(H_FEATS * IN_FEATS + T - 1) / T, T, 0, stream>>>(W1, W1t, IN_FEATS, H_FEATS, H_FEATS);
  k_transpose_bf16<<<(CAT * CAT + T - 1) / T, T, 0, stream>>>(l1w, l1t, CAT, CAT, CAT);
  k_transpose_bf16<<<(CAT * CAT + T - 1) / T, T, 0, stream>>>(l2w, l2t, CAT, CAT, CAT);
  k_transpose_bf16<<<(H_FEATS * CAT + T - 1) / T, T, 0, stream>>>(W2, W2t, CAT, H_FEATS, H_FEATS);
  k_transpose_bf16<<<(64 * CAT + T - 1) / T, T, 0, stream>>>(l3w, l3t, CAT, NUM_CLASSES, 64);

  auto gemmGrid = [](int Nt64) { int waves = (N_NODES / 16) * Nt64; return (waves + 7) / 8; };

  // --- gconv layer 1: Y = X @ W1 (shared across graphs; deg_out scale folded into edges) ---
  k_gemm_wmma<false, false, false><<<gemmGrid(2), 256, 0, stream>>>(
      Xb, W1t, nullptr, Y, N_NODES, IN_FEATS, 2, H_FEATS, H_FEATS);
  for (int g = 0; g < G_NUM; ++g) {
    k_zero_f32<<<(nagg + T - 1) / T, T, 0, stream>>>(agg, nagg);
    k_spmm<<<(N_EDGES + 7) / 8, 256, 0, stream>>>(
        src + (size_t)g * N_EDGES, dst + (size_t)g * N_EDGES, w + (size_t)g * N_EDGES,
        degout + g * N_NODES, Y, agg, N_EDGES);
    k_finalize<<<(nagg + T - 1) / T, T, 0, stream>>>(agg, degin + g * N_NODES, bufA, g * H_FEATS);
  }

  // --- dense MLP: h = relu(h @ l1w + b) ; h = relu(h @ l2w + b) ---
  k_gemm_wmma<true, true, true><<<gemmGrid(8), 256, 0, stream>>>(
      bufA, l1t, l1b, bufB, N_NODES, CAT, 8, CAT, CAT);
  k_gemm_wmma<true, true, true><<<gemmGrid(8), 256, 0, stream>>>(
      bufB, l2t, l2b, bufA, N_NODES, CAT, 8, CAT, CAT);

  // --- gconv layer 2: Z = H2 @ W2 ---
  k_gemm_wmma<false, false, false><<<gemmGrid(2), 256, 0, stream>>>(
      bufA, W2t, nullptr, Y, N_NODES, CAT, 2, H_FEATS, H_FEATS);
  for (int g = 0; g < G_NUM; ++g) {
    k_zero_f32<<<(nagg + T - 1) / T, T, 0, stream>>>(agg, nagg);
    k_spmm<<<(N_EDGES + 7) / 8, 256, 0, stream>>>(
        src + (size_t)g * N_EDGES, dst + (size_t)g * N_EDGES, w + (size_t)g * N_EDGES,
        degout + g * N_NODES, Y, agg, N_EDGES);
    k_finalize<<<(nagg + T - 1) / T, T, 0, stream>>>(agg, degin + g * N_NODES, bufB, g * H_FEATS);
  }

  // --- classifier: out = C2 @ l3w + b (f32 -> d_out) ---
  k_gemm_wmma<false, false, true><<<gemmGrid(1), 256, 0, stream>>>(
      bufB, l3t, l3b, d_out, N_NODES, CAT, 1, NUM_CLASSES, NUM_CLASSES);
}
